// MultiHeadAttention_4836133175991
// MI455X (gfx1250) — matrix operand order repaired and checked by reading, not compiled, not run on hardware
//
#include <hip/hip_runtime.h>
#include <hip/hip_bf16.h>

// ---------------------------------------------------------------------------
// CDNA5 (gfx1250) fused attention. wave32, v_wmma_f32_16x16x32_bf16 +
// TDM (tensor_load_to_lds) staging of hq/hk tiles.
// B=2, T=256, C=128, H=4, HS=32.
// ---------------------------------------------------------------------------

typedef __attribute__((ext_vector_type(16))) __bf16        v16bf;
typedef __attribute__((ext_vector_type(8)))  __bf16        v8bf;
typedef __attribute__((ext_vector_type(8)))  float         v8f;
typedef __attribute__((ext_vector_type(8)))  unsigned int  v8u;
typedef __attribute__((ext_vector_type(4)))  unsigned int  v4u;
typedef __attribute__((ext_vector_type(8)))  int           v8i;
typedef __attribute__((ext_vector_type(4)))  int           v4i;

#define Bb 2
#define Tt 256
#define Cc 128
#define Hh 4
#define HSs 32

__device__ __forceinline__ __bf16 f2bf(float f) {      // RNE (prep path)
    unsigned int u = __float_as_uint(f);
    unsigned short h = (unsigned short)((u + 0x7FFFu + ((u >> 16) & 1u)) >> 16);
    return __builtin_bit_cast(__bf16, h);
}

// Pack two fp32 into dword of two bf16 (truncation) with one v_perm_b32.
__device__ __forceinline__ unsigned int pk_bf(float lo, float hi) {
    return __builtin_amdgcn_perm(__float_as_uint(hi), __float_as_uint(lo),
                                 0x07060302u);
}

__device__ __forceinline__ float gelu_exact(float x) {
    return 0.5f * x * (1.0f + erff(x * 0.70710678118654752440f));
}

__device__ __forceinline__ v8f vzero8() {
    v8f z;
#pragma unroll
    for (int i = 0; i < 8; ++i) z[i] = 0.0f;
    return z;
}

// Concatenate two 8-element bf16 vectors into one WMMA fragment register set.
__device__ __forceinline__ v16bf cat8(v8bf a, v8bf b) {
    v4u ua = __builtin_bit_cast(v4u, a);
    v4u ub = __builtin_bit_cast(v4u, b);
    v8u r;
#pragma unroll
    for (int i = 0; i < 4; ++i) { r[i] = ua[i]; r[i + 4] = ub[i]; }
    return __builtin_bit_cast(v16bf, r);
}

// A fragment (16x32 bf16, MxK) per ISA 7.12.2: lane holds one M row
// (row = lane&15); k = kb+{0..7} and kb+16+{0..7}, kb = ksBase + 8*(lane>>4).
__device__ __forceinline__ v16bf load_a_bf16(const __bf16* row, int kb) {
    v8bf r0 = *(const v8bf*)(row + kb);
    v8bf r1 = *(const v8bf*)(row + kb + 16);
    return cat8(r0, r1);
}

// Same A fragment from fp32 memory (pos_dist_emb): 4x global_load_b128 +
// 8x v_perm_b32 (truncate-to-bf16 packing).
__device__ __forceinline__ v16bf load_a_f32(const float* row, int kb) {
    float4 a0 = *(const float4*)(row + kb);
    float4 a1 = *(const float4*)(row + kb + 4);
    float4 a2 = *(const float4*)(row + kb + 16);
    float4 a3 = *(const float4*)(row + kb + 20);
    v8u r;
    r[0] = pk_bf(a0.x, a0.y); r[1] = pk_bf(a0.z, a0.w);
    r[2] = pk_bf(a1.x, a1.y); r[3] = pk_bf(a1.z, a1.w);
    r[4] = pk_bf(a2.x, a2.y); r[5] = pk_bf(a2.z, a2.w);
    r[6] = pk_bf(a3.x, a3.y); r[7] = pk_bf(a3.z, a3.w);
    return __builtin_bit_cast(v16bf, r);
}

// B fragment (32x16 bf16, KxN) from TRANSPOSED [n][k] storage: lane holds
// column n (pre-offset row pointer), k = kb8+{0..7} and kb8+16+{0..7},
// kb8 = ksBase + 8*(lane>>4) (caller passes ksBase + 16*(lane>>4)). Two _b128 loads.
__device__ __forceinline__ v16bf load_b_t(const __bf16* row, int kb) {
    v8bf r0 = *(const v8bf*)(row + kb - ((kb & 16) >> 1));
    v8bf r1 = *(const v8bf*)(row + kb - ((kb & 16) >> 1) + 16);
    return cat8(r0, r1);
}

__device__ __forceinline__ v8f wmma_bf16(v16bf a, v16bf b, v8f c) {
    return __builtin_amdgcn_wmma_f32_16x16x32_bf16(false, a, false, b,
                                                   (short)0, c, false, false);
}

// ---------------------------------------------------------------------------
// TDM: DMA a 2D fp32 tile (rows x cols, row stride in elements) from global
// memory into LDS. Descriptor per CDNA5 ISA 8.3/8.4:
//   g0: [1:0]=count=1 | lds_addr[63:32] | global_addr[120:64] | type=2
//   g1: data_size=2(4B) | tensor_dim0/1 | tile_dim0/1 | tensor_dim0_stride
// Tracked by TENSORcnt (S_WAIT_TENSORCNT). EXEC ignored; issue from one wave.
// ---------------------------------------------------------------------------
__device__ __forceinline__ void tdm_load_2d_f32(unsigned lds_off,
                                                const float* gptr,
                                                unsigned rows, unsigned cols,
                                                unsigned row_stride) {
    unsigned long long ga = (unsigned long long)gptr;
    v4u g0;
    g0[0] = 1u;                                          // count=1 (user D#)
    g0[1] = lds_off;                                     // lds_addr (bytes)
    g0[2] = (unsigned)(ga & 0xFFFFFFFFu);                // global_addr lo
    g0[3] = (unsigned)((ga >> 32) & 0x01FFFFFFu)         // global_addr hi
          | (2u << 30);                                  // type = 2 ("image")
    v8i g1;
    g1[0] = (int)(2u << 16);                             // data_size=4B
    g1[1] = (int)(cols << 16);                           // tensor_dim0 lo16
    g1[2] = (int)((cols >> 16) | (rows << 16));          // dim0 hi | dim1 lo
    g1[3] = (int)((rows >> 16) | (cols << 16));          // dim1 hi | tile_dim0
    g1[4] = (int)rows;                                   // tile_dim1 (dim2=0)
    g1[5] = (int)row_stride;                             // dim0_stride lo32
    g1[6] = 0;
    g1[7] = 0;
    v4i gz = {0, 0, 0, 0};
#if __clang_major__ >= 23
    v8i gz8 = {0, 0, 0, 0, 0, 0, 0, 0};
    __builtin_amdgcn_tensor_load_to_lds(g0, g1, gz, gz, gz8, 0);
#else
    __builtin_amdgcn_tensor_load_to_lds(g0, g1, gz, gz, 0);
#endif
}

__device__ __forceinline__ unsigned lds_offset(const void* p) {
    // LDS aperture: flat addr[31:0] == LDS byte offset (ISA 10.2).
    return (unsigned)(unsigned long long)p;
}

// ---------------------------------------------------------------------------
// Kernel 0: bf16 conversion + weight transpose.
//   x1b[bt][c] = x + st_pos_emb          (65536)
//   xb [bt][c] = x                       (65536)
//   w1t[h][d][c] = W1[h][c][d]           (262144)   c<512: k|q|p rows
//   wvt[h][e][c] = Wv[h][c][e]           (16384)
// ---------------------------------------------------------------------------
__global__ void prep_kernel(const float* __restrict__ x,
                            const float* __restrict__ spe,
                            const float* __restrict__ W1,
                            const float* __restrict__ Wv,
                            __bf16* __restrict__ x1b, __bf16* __restrict__ xb,
                            __bf16* __restrict__ w1t, __bf16* __restrict__ wvt) {
    int step = gridDim.x * blockDim.x;
    for (int idx = blockIdx.x * blockDim.x + threadIdx.x; idx < 409600; idx += step) {
        if (idx < 65536) {
            x1b[idx] = f2bf(x[idx] + spe[idx & 32767]);        // T*C = 32768
        } else if (idx < 131072) {
            int j = idx - 65536;  xb[j] = f2bf(x[j]);
        } else if (idx < 393216) {
            int j = idx - 131072;                 // j = (h*128 + d)*512 + c
            int c = j & 511, hd = j >> 9;
            int d = hd & 127, h = hd >> 7;
            w1t[j] = f2bf(W1[(h * 512 + c) * 128 + d]);
        } else {
            int j = idx - 393216;                 // j = (h*32 + e)*128 + c
            int c = j & 127, he = j >> 7;
            int e = he & 31, h = he >> 5;
            wvt[j] = f2bf(Wv[(h * 128 + c) * 32 + e]);
        }
    }
}

// ---------------------------------------------------------------------------
// Kernel 1: hq = x1 @ W1q + b1 ; hk = x1 @ W1k ; v = x @ Wv + bv.
// grid = H * (B*T/16) = 128 blocks, 256 threads (8 waves).
// ---------------------------------------------------------------------------
__global__ __launch_bounds__(256, 2)
void qkv_kernel(const __bf16* __restrict__ x1b, const __bf16* __restrict__ xb,
                const __bf16* __restrict__ w1t, const __bf16* __restrict__ wvt,
                const float* __restrict__ b1, const float* __restrict__ bv,
                float* __restrict__ hq, float* __restrict__ hk,
                __bf16* __restrict__ vt) {
    const int bid  = blockIdx.x;
    const int h    = bid >> 5;
    const int m0   = (bid & 31) << 4;
    const int tid  = threadIdx.x;
    const int w    = tid >> 5;
    const int lane = tid & 31;
    const int hi   = lane >> 4;
    const int lo   = lane & 15;
    const int d0   = w << 4;

    const __bf16* wqr = w1t + (h * 128 + d0 + lo) * 512 + 128;  // W1q rows
    const __bf16* wkr = w1t + (h * 128 + d0 + lo) * 512;        // W1k rows
    const __bf16* arow = x1b + (m0 + lo) * Cc;

    v8f accq, acck = vzero8();
    {
        float bqv = b1[h * Cc + d0 + lo];
#pragma unroll
        for (int r = 0; r < 8; ++r) accq[r] = bqv;
    }

#pragma unroll
    for (int ks = 0; ks < 4; ++ks) {
        v16bf a  = load_a_bf16(arow, ks * 32 + 8 * hi);
        v16bf bq = load_b_t(wqr, ks * 32 + 16 * hi);
        accq = wmma_bf16(a, bq, accq);
        v16bf bk = load_b_t(wkr, ks * 32 + 16 * hi);
        acck = wmma_bf16(a, bk, acck);
    }
#pragma unroll
    for (int r = 0; r < 8; ++r) {
        int m = m0 + r + 8 * hi;
        hq[(h * 512 + m) * Cc + d0 + lo] = accq[r];
        hk[(h * 512 + m) * Cc + d0 + lo] = acck[r];
    }

    if (w < 2) {   // v projection, N = 32
        const int e0 = w << 4;
        v8f accv;
        float bvv = bv[h * HSs + e0 + lo];
#pragma unroll
        for (int r = 0; r < 8; ++r) accv[r] = bvv;
        const __bf16* arow2 = xb + (m0 + lo) * Cc;
        const __bf16* wvr = wvt + (h * HSs + e0 + lo) * Cc;
#pragma unroll
        for (int ks = 0; ks < 4; ++ks) {
            v16bf a   = load_a_bf16(arow2, ks * 32 + 8 * hi);
            v16bf bvf = load_b_t(wvr, ks * 32 + 16 * hi);
            accv = wmma_bf16(a, bvf, accv);
        }
        // Store transposed: vt[((h*B + b)*HS + e)*T + s]
#pragma unroll
        for (int r = 0; r < 8; ++r) {
            int m = m0 + r + 8 * hi;
            int bsel = m >> 8, s = m & 255;
            vt[(((h * Bb + bsel) * HSs) + e0 + lo) * Tt + s] = f2bf(accv[r]);
        }
    }
}

// ---------------------------------------------------------------------------
// Kernel 2: fused scores + softmax + P@V.
// grid = B*H*(T/16) = 128 blocks, 256 threads (8 waves).
// hq/hk tiles staged into LDS by the Tensor Data Mover (TENSORcnt-tracked),
// waited on before each s-tile's WMMAs; pos_dist_emb streamed from L2 as WMMA-A.
// ---------------------------------------------------------------------------
__global__ __launch_bounds__(256, 1)
void attn_kernel(const float* __restrict__ pde, const __bf16* __restrict__ w1t,
                 const float* __restrict__ W2, const float* __restrict__ b2,
                 const float* __restrict__ hq, const float* __restrict__ hk,
                 const __bf16* __restrict__ vt, float* __restrict__ out) {
    __shared__ float  hq_s[16 * Cc];        // 8 KB  (TDM destination)
    __shared__ float  hk_s[16 * Cc];        // 8 KB  (TDM destination)
    __shared__ float  wei_s[16 * Tt];       // 16 KB
    __shared__ __bf16 p_s[16 * Tt];         // 8 KB

    const int bid  = blockIdx.x;
    const int tile = bid & 15;
    const int h    = (bid >> 4) & 3;
    const int b    = bid >> 6;
    const int t0   = tile << 4;
    const int tid  = threadIdx.x;
    const int w    = tid >> 5;
    const int lane = tid & 31;
    const int hi   = lane >> 4;
    const int lo   = lane & 15;
    const int pb   = w << 5;                // pair base for this wave

    // TDM: stage the hq tile once (16 x 128 fp32, row stride 128).
    if (tid < 32) {
        const float* hqp = hq + (h * 512 + b * Tt + t0) * Cc;
        tdm_load_2d_f32(lds_offset(hq_s), hqp, 16, Cc, Cc);
    }

    const float* w2p = W2 + h * Cc;

    for (int st = 0; st < 16; ++st) {
        __syncthreads();                    // everyone done with old hk_s
        if (tid < 32) {
            const float* hkp = hk + (h * 512 + b * Tt + st * 16) * Cc;
            tdm_load_2d_f32(lds_offset(hk_s), hkp, 16, Cc, Cc);
            __builtin_amdgcn_s_wait_tensorcnt(0);   // hq (st==0) + hk done
        }
        __syncthreads();

        float wacc[2][8];
#pragma unroll
        for (int mt = 0; mt < 2; ++mt)
#pragma unroll
            for (int r = 0; r < 8; ++r) wacc[mt][r] = 0.0f;

        // Per-lane A row pointers (two 16-row M subtiles of (t,s) pairs).
        const float* arow[2];
#pragma unroll
        for (int mt = 0; mt < 2; ++mt) {
            int pair = pb + mt * 16 + lo;
            int tl = pair >> 4, sl = pair & 15;
            arow[mt] = pde + (((b * Tt + t0 + tl) * Tt) + st * 16 + sl) * (2 * Cc);
        }
        __builtin_prefetch(arow[0], 0, 0);          // global_prefetch_b8
        __builtin_prefetch(arow[1], 0, 0);

        for (int nh = 0; nh < 2; ++nh) {            // two 64-wide N halves
            const int dn0 = nh * 64;
            const __bf16* brow[4];
#pragma unroll
            for (int nt = 0; nt < 4; ++nt)
                brow[nt] = w1t + (h * 128 + dn0 + nt * 16 + lo) * 512 + 256;

            v8f acc[2][4];
#pragma unroll
            for (int mt = 0; mt < 2; ++mt)
#pragma unroll
                for (int nt = 0; nt < 4; ++nt) acc[mt][nt] = vzero8();

#pragma unroll
            for (int ks = 0; ks < 8; ++ks) {        // K = 256 -> 8 x 32
                int kb = ks * 32 + 8 * hi;
                v16bf a0 = load_a_f32(arow[0], kb);
                v16bf a1 = load_a_f32(arow[1], kb);
#pragma unroll
                for (int nt = 0; nt < 4; ++nt) {
                    v16bf bb = load_b_t(brow[nt], ks * 32 + 16 * hi);
                    acc[0][nt] = wmma_bf16(a0, bb, acc[0][nt]);
                    acc[1][nt] = wmma_bf16(a1, bb, acc[1][nt]);
                }
            }

            // Epilogue: + hq + hk (b1 folded into hq), GELU, * W2, reduce d.
#pragma unroll
            for (int mt = 0; mt < 2; ++mt) {
#pragma unroll
                for (int nt = 0; nt < 4; ++nt) {
#pragma unroll
                    for (int r = 0; r < 8; ++r) {
                        int pair = pb + mt * 16 + r + 8 * hi;
                        int tl = pair >> 4, sl = pair & 15;
                        int d = dn0 + nt * 16 + lo;
                        float v = acc[mt][nt][r] + hq_s[tl * Cc + d] + hk_s[sl * Cc + d];
                        v = gelu_exact(v) * w2p[d];
                        v += __shfl_xor(v, 1);
                        v += __shfl_xor(v, 2);
                        v += __shfl_xor(v, 4);
                        v += __shfl_xor(v, 8);      // sum over 16 d-lanes
                        wacc[mt][r] += v;
                    }
                }
            }
        }

        if (lo == 0) {
#pragma unroll
            for (int mt = 0; mt < 2; ++mt)
#pragma unroll
                for (int r = 0; r < 8; ++r) {
                    int pair = pb + mt * 16 + r + 8 * hi;
                    int tl = pair >> 4, sl = pair & 15;
                    wei_s[tl * Tt + st * 16 + sl] = wacc[mt][r];
                }
        }
    }
    __syncthreads();

    // Masked softmax over s (rows 2w, 2w+1 per wave; lane-parallel over 256).
    const float scale = 0.08838834764831845f;   // 128^-0.5
    const float b2h = b2[h];
#pragma unroll
    for (int rr = 0; rr < 2; ++rr) {
        int tl = w * 2 + rr;
        int tg = t0 + tl;
        float vals[8];
        float m = -3.4e38f;
#pragma unroll
        for (int j = 0; j < 8; ++j) {
            int s = lane + j * 32;
            float xv = (wei_s[tl * Tt + s] + b2h) * scale;
            if (s > tg) xv = -1e30f;
            vals[j] = xv;
            m = fmaxf(m, xv);
        }
        m = fmaxf(m, __shfl_xor(m, 1));
        m = fmaxf(m, __shfl_xor(m, 2));
        m = fmaxf(m, __shfl_xor(m, 4));
        m = fmaxf(m, __shfl_xor(m, 8));
        m = fmaxf(m, __shfl_xor(m, 16));
        float sum = 0.0f;
#pragma unroll
        for (int j = 0; j < 8; ++j) {
            float e = __expf(vals[j] - m);
            vals[j] = e;
            sum += e;
        }
        sum += __shfl_xor(sum, 1);
        sum += __shfl_xor(sum, 2);
        sum += __shfl_xor(sum, 4);
        sum += __shfl_xor(sum, 8);
        sum += __shfl_xor(sum, 16);
        float inv = 1.0f / sum;
#pragma unroll
        for (int j = 0; j < 8; ++j)
            p_s[tl * Tt + lane + j * 32] = f2bf(vals[j] * inv);
    }
    __syncthreads();

    // out[b, t, h*32+e] = P(16x256) @ V(256x32). Waves 0..1, e0 = 16*w.
    if (w < 2) {
        const int e0 = w << 4;
        const __bf16* vrow = vt + (((h * Bb + b) * HSs) + e0 + lo) * Tt;
        v8f acc = vzero8();
#pragma unroll
        for (int ks = 0; ks < 8; ++ks) {
            int kb = ks * 32 + 8 * hi;
            v16bf a  = load_a_bf16(p_s + lo * Tt, kb);
            v16bf bb = load_b_t(vrow, ks * 32 + 16 * hi);
            acc = wmma_bf16(a, bb, acc);
        }
#pragma unroll
        for (int r = 0; r < 8; ++r) {
            int tl = r + 8 * hi;
            out[(b * Tt + t0 + tl) * (Hh * HSs) + h * HSs + e0 + lo] = acc[r];
        }
    }
}

// ---------------------------------------------------------------------------
// Launch. ws layout (bytes):
//   x1b @0 (128K) | xb @128K (128K) | w1t @256K (512K) | wvt @768K (32K)
//   hq @800K (1M) | hk @1824K (1M) | vt @2848K (128K)   total < 3 MB
// ---------------------------------------------------------------------------
extern "C" void kernel_launch(void* const* d_in, const int* in_sizes, int n_in,
                              void* d_out, int out_size, void* d_ws, size_t ws_size,
                              hipStream_t stream) {
    (void)in_sizes; (void)n_in; (void)out_size; (void)ws_size;
    const float* x   = (const float*)d_in[0];
    const float* spe = (const float*)d_in[1];
    const float* pde = (const float*)d_in[2];
    const float* W1  = (const float*)d_in[3];
    const float* b1  = (const float*)d_in[4];
    const float* W2  = (const float*)d_in[5];
    const float* b2  = (const float*)d_in[6];
    const float* Wv  = (const float*)d_in[7];
    const float* bv  = (const float*)d_in[8];
    float* out = (float*)d_out;

    char* ws = (char*)d_ws;
    __bf16* x1b = (__bf16*)(ws + 0);
    __bf16* xb  = (__bf16*)(ws + (128u << 10));
    __bf16* w1t = (__bf16*)(ws + (256u << 10));
    __bf16* wvt = (__bf16*)(ws + (768u << 10));
    float*  hq  = (float*)(ws + (800u << 10));
    float*  hk  = (float*)(ws + (1824u << 10));
    __bf16* vt  = (__bf16*)(ws + (2848u << 10));

    prep_kernel<<<1600, 256, 0, stream>>>(x, spe, W1, Wv, x1b, xb, w1t, wvt);
    qkv_kernel<<<128, 256, 0, stream>>>(x1b, xb, w1t, wvt, b1, bv, hq, hk, vt);
    attn_kernel<<<128, 256, 0, stream>>>(pde, w1t, W2, b2, hq, hk, vt, out);
}
